// DGAT_88802743812894
// MI455X (gfx1250) — matrix operand order repaired and checked by reading, not compiled, not run on hardware
//
#include <hip/hip_runtime.h>
#include <hip/hip_bf16.h>

typedef __attribute__((ext_vector_type(8)))  _Float16 v8h;
typedef __attribute__((ext_vector_type(16))) _Float16 v16h;
typedef __attribute__((ext_vector_type(8)))  float    v8f;

#define NN 10000
#define HH 32
#define EE 80000
#define EP 90000   // E + N self loops

// ---------------------------------------------------------------- utilities

__device__ __forceinline__ void atomicMaxFloat(float* addr, float val) {
  // float ordering via int (>=0) / uint (<0) atomics; init value is -1e30
  if (val >= 0.f) atomicMax((int*)addr, __float_as_int(val));
  else            atomicMin((unsigned int*)addr, __float_as_uint(val));
}

__global__ void k_f32_to_f16(_Float16* __restrict__ dst, const float* __restrict__ src, int n) {
  int i = blockIdx.x * blockDim.x + threadIdx.x;
  if (i < n) dst[i] = (_Float16)src[i];
}

__global__ void k_fill(float* __restrict__ p, float v, int n) {
  int i = blockIdx.x * blockDim.x + threadIdx.x;
  if (i < n) p[i] = v;
}

// W [K, Nc] f32 -> Wt [Nc, K] f16 (so B fragments are contiguous in K)
__global__ void k_transpose_to_f16(_Float16* __restrict__ dst, const float* __restrict__ src,
                                   int K, int Nc) {
  int i = blockIdx.x * blockDim.x + threadIdx.x;
  if (i >= K * Nc) return;
  int k = i / Nc, n = i - k * Nc;
  dst[(size_t)n * K + k] = (_Float16)src[i];
}

// ------------------------------------------------------------- WMMA GEMM
// C[M,Nc] = A[M,K] (f16, row major) @ Bt[Nc,K]^T (f16, K-contiguous)
// Each wave32 computes a 16 x (16*NT) strip: the A fragment is loaded once
// per k-step and reused across NT v_wmma issues (raises WMMA/VMEM density).
// K multiple of 32, M multiple of 16, Nc multiple of 16*NT.
template <int NT>
__global__ void k_gemm_wmma(const _Float16* __restrict__ A, const _Float16* __restrict__ Bt,
                            float* __restrict__ C, int M, int K, int Nc) {
  const int ntiles  = Nc >> 4;
  const int nstrips = ntiles / NT;
  int wave  = blockIdx.x * (blockDim.x >> 5) + (threadIdx.x >> 5);
  int total = (M >> 4) * nstrips;
  if (wave >= total) return;                 // wave-uniform: EXEC stays all-1s
  int lane = threadIdx.x & 31;
  int tm = wave / nstrips, ts = wave - tm * nstrips;
  int l16 = lane & 15, hi = lane >> 4;
  const _Float16* arow = A + (size_t)(tm * 16 + l16) * K;
  const _Float16* brow[NT];
#pragma unroll
  for (int t = 0; t < NT; ++t)
    brow[t] = Bt + (size_t)((ts * NT + t) * 16 + l16) * K;
  v8f acc[NT];
#pragma unroll
  for (int t = 0; t < NT; ++t) acc[t] = (v8f){};
  for (int k0 = 0; k0 < K; k0 += 32) {
    if (k0 + 64 < K)                         // gfx1250 global_prefetch_b8 path
      __builtin_prefetch(arow + k0 + 64, 0, 1);
    // A 16x32 f16 layout: lane<16 -> K {k0..k0+7, k0+16..k0+23}; lane>=16 +8
    int ka = k0 + hi * 8;
    v8h a0 = *(const v8h*)(arow + ka);
    v8h a1 = *(const v8h*)(arow + ka + 16);
    v16h av = __builtin_shufflevector(a0, a1, 0,1,2,3,4,5,6,7,8,9,10,11,12,13,14,15);
    // B 32x16 f16 layout: lane<16 -> K {k0..k0+7, k0+16..k0+23}; lane>=16 +8
    int kb = k0 + hi * 8;
#pragma unroll
    for (int t = 0; t < NT; ++t) {
      v8h b0 = *(const v8h*)(brow[t] + kb);
      v8h b1 = *(const v8h*)(brow[t] + kb + 16);
      v16h bv = __builtin_shufflevector(b0, b1, 0,1,2,3,4,5,6,7,8,9,10,11,12,13,14,15);
      acc[t] = __builtin_amdgcn_wmma_f32_16x16x32_f16(false, av, false, bv, (short)0,
                                                      acc[t], false, false);
    }
  }
#pragma unroll
  for (int t = 0; t < NT; ++t) {
    int col = (ts * NT + t) * 16 + l16;
#pragma unroll
    for (int r = 0; r < 8; ++r) {
      int row = tm * 16 + r + hi * 8;        // D: VGPR r -> M = r + 8*(lane>=16)
      C[(size_t)row * Nc + col] = acc[t][r];
    }
  }
}

// ------------------------------------------------------------- GAT kernels

// es/ed[i,h] = sum_c h[i,h,c] * a_{s,d}[h,c]
__global__ void k_node_logits(const float* __restrict__ hmat, const float* __restrict__ a_s,
                              const float* __restrict__ a_d, float* __restrict__ es,
                              float* __restrict__ ed, int C) {
  int gid = blockIdx.x * blockDim.x + threadIdx.x;
  if (gid >= NN * HH) return;
  int i = gid >> 5, hd = gid & 31;
  const float* hp = hmat + ((size_t)i * HH + hd) * C;
  const float* as = a_s + hd * C;
  const float* ad = a_d + hd * C;
  float s = 0.f, d = 0.f;
  for (int c = 0; c < C; ++c) { float v = hp[c]; s += v * as[c]; d += v * ad[c]; }
  es[gid] = s; ed[gid] = d;
}

__device__ __forceinline__ void edge_ends(int e, const int* ei, int flip, int& src, int& dst) {
  if (e < EE) {
    src = ei[flip ? EE + e : e];
    dst = ei[flip ? e : EE + e];
  } else {
    src = dst = e - EE;                      // self loops
  }
}

// pass 1: e = leaky_relu(es[src]+ed[dst]); store; segment max into m[dst]
__global__ void k_edge_logits(const float* __restrict__ es, const float* __restrict__ ed,
                              const int* __restrict__ ei, float* __restrict__ m,
                              float* __restrict__ p, int flip) {
  int gid = blockIdx.x * blockDim.x + threadIdx.x;
  if (gid >= EP * HH) return;
  int e = gid >> 5, hd = gid & 31, src, dst;
  edge_ends(e, ei, flip, src, dst);
  float v = es[src * HH + hd] + ed[dst * HH + hd];
  v = v > 0.f ? v : 0.2f * v;
  p[gid] = v;
  atomicMaxFloat(&m[dst * HH + hd], v);
}

// pass 2: p = exp(e - m[dst]); segment sum into z[dst]
__global__ void k_edge_exp(float* __restrict__ p, const float* __restrict__ m,
                           const int* __restrict__ ei, float* __restrict__ z, int flip) {
  int gid = blockIdx.x * blockDim.x + threadIdx.x;
  if (gid >= EP * HH) return;
  int e = gid >> 5, hd = gid & 31, src, dst;
  edge_ends(e, ei, flip, src, dst);
  float v = expf(p[gid] - m[dst * HH + hd]);
  p[gid] = v;
  atomicAdd(&z[dst * HH + hd], v);
}

// pass 3: alpha = p / z[dst]
__global__ void k_edge_alpha(float* __restrict__ p, const float* __restrict__ z,
                             const int* __restrict__ ei, int flip) {
  int gid = blockIdx.x * blockDim.x + threadIdx.x;
  if (gid >= EP * HH) return;
  int e = gid >> 5, hd = gid & 31, src, dst;
  edge_ends(e, ei, flip, src, dst);
  p[gid] = p[gid] / z[dst * HH + hd];
}

// pass 4: agg[dst,c] += (1/H) * sum_h alpha[e,h] * h[src,h,c]   (head-mean folded in)
__global__ void k_edge_agg(const float* __restrict__ p, const float* __restrict__ hmat,
                           const int* __restrict__ ei, float* __restrict__ agg,
                           int C, int flip) {
  int gid = blockIdx.x * blockDim.x + threadIdx.x;
  if (gid >= EP * C) return;
  int e = gid / C, c = gid - e * C, src, dst;
  edge_ends(e, ei, flip, src, dst);
  const float* pe = p + (size_t)e * HH;
  const float* hs = hmat + (size_t)src * HH * C + c;
  float acc = 0.f;
#pragma unroll
  for (int hd = 0; hd < HH; ++hd) acc += pe[hd] * hs[(size_t)hd * C];
  atomicAdd(&agg[dst * C + c], acc * (1.f / 32.f));
}

// out = elu(agg + bias)
__global__ void k_gat_out(const float* __restrict__ agg, const float* __restrict__ bias,
                          float* __restrict__ out, int C) {
  int gid = blockIdx.x * blockDim.x + threadIdx.x;
  if (gid >= NN * C) return;
  int c = gid % C;
  float v = agg[gid] + bias[c];
  out[gid] = v > 0.f ? v : (expf(v) - 1.f);
}

// ---------------------------------------------------------- residual block

__global__ void k_mm16(const float* __restrict__ A, const float* __restrict__ W,
                       const float* __restrict__ bias, float* __restrict__ out) {
  int gid = blockIdx.x * blockDim.x + threadIdx.x;
  if (gid >= NN * 16) return;
  int i = gid >> 4, j = gid & 15;
  float acc = bias[j];
  const float* ar = A + i * 16;
#pragma unroll
  for (int k = 0; k < 16; ++k) acc += ar[k] * W[k * 16 + j];
  out[gid] = acc;
}

__global__ void k_bias_bn(const float* __restrict__ t, const float* __restrict__ bias,
                          const float* __restrict__ g, const float* __restrict__ bb,
                          const float* __restrict__ mm, const float* __restrict__ vv,
                          float* __restrict__ out, int hasbias, int dorelu) {
  int gid = blockIdx.x * blockDim.x + threadIdx.x;
  if (gid >= NN * 16) return;
  int j = gid & 15;
  float v = t[gid] + (hasbias ? bias[j] : 0.f);
  v = (v - mm[j]) * g[j] / sqrtf(vv[j] + 1e-5f) + bb[j];
  if (dorelu) v = fmaxf(v, 0.f);
  out[gid] = v;
}

__global__ void k_add_relu(const float* __restrict__ a, const float* __restrict__ b,
                           float* __restrict__ out, int n) {
  int i = blockIdx.x * blockDim.x + threadIdx.x;
  if (i < n) out[i] = fmaxf(a[i] + b[i], 0.f);
}

// ---------------------------------------------------------------- driver

static inline int gdim(int n, int b) { return (n + b - 1) / b; }

static void run_gat_layer(const _Float16* Ah, int K, const float* W, const float* a_s,
                          const float* a_d, const float* bias, int C, const int* ei, int flip,
                          _Float16* Wt, float* h, float* es, float* ed, float* m, float* z,
                          float* p, float* agg, float* out, hipStream_t stream) {
  const int HC = HH * C;
  k_transpose_to_f16<<<gdim(K * HC, 256), 256, 0, stream>>>(Wt, W, K, HC);
  int waves = (NN / 16) * (HC / 64);         // NT=4 strips (HC is 2048 or 512)
  k_gemm_wmma<4><<<gdim(waves, 4), 128, 0, stream>>>(Ah, Wt, h, NN, K, HC);
  k_node_logits<<<gdim(NN * HH, 256), 256, 0, stream>>>(h, a_s, a_d, es, ed, C);
  k_fill<<<gdim(NN * HH, 256), 256, 0, stream>>>(m, -1e30f, NN * HH);
  k_fill<<<gdim(NN * HH, 256), 256, 0, stream>>>(z, 0.f, NN * HH);
  k_fill<<<gdim(NN * C, 256), 256, 0, stream>>>(agg, 0.f, NN * C);
  k_edge_logits<<<gdim(EP * HH, 256), 256, 0, stream>>>(es, ed, ei, m, p, flip);
  k_edge_exp<<<gdim(EP * HH, 256), 256, 0, stream>>>(p, m, ei, z, flip);
  k_edge_alpha<<<gdim(EP * HH, 256), 256, 0, stream>>>(p, z, ei, flip);
  k_edge_agg<<<gdim(EP * C, 256), 256, 0, stream>>>(p, h, ei, agg, C, flip);
  k_gat_out<<<gdim(NN * C, 256), 256, 0, stream>>>(agg, bias, out, C);
}

extern "C" void kernel_launch(void* const* d_in, const int* in_sizes, int n_in,
                              void* d_out, int out_size, void* d_ws, size_t ws_size,
                              hipStream_t stream) {
  const float* x   = (const float*)d_in[0];
  const int*   ei  = (const int*)  d_in[1];
  const float* W1  = (const float*)d_in[2];
  const float* as1 = (const float*)d_in[3];
  const float* ad1 = (const float*)d_in[4];
  const float* b1  = (const float*)d_in[5];
  const float* W2  = (const float*)d_in[6];
  const float* as2 = (const float*)d_in[7];
  const float* ad2 = (const float*)d_in[8];
  const float* b2  = (const float*)d_in[9];
  const float* W3  = (const float*)d_in[10];
  const float* as3 = (const float*)d_in[11];
  const float* ad3 = (const float*)d_in[12];
  const float* b3  = (const float*)d_in[13];
  const float* W4  = (const float*)d_in[14];
  const float* as4 = (const float*)d_in[15];
  const float* ad4 = (const float*)d_in[16];
  const float* b4  = (const float*)d_in[17];
  const float* rwa = (const float*)d_in[18];
  const float* rba = (const float*)d_in[19];
  const float* bn1g = (const float*)d_in[20];
  const float* bn1b = (const float*)d_in[21];
  const float* bn1m = (const float*)d_in[22];
  const float* bn1v = (const float*)d_in[23];
  const float* rwb  = (const float*)d_in[24];
  const float* rbb  = (const float*)d_in[25];
  const float* bn2g = (const float*)d_in[26];
  const float* bn2b = (const float*)d_in[27];
  const float* bn2m = (const float*)d_in[28];
  const float* bn2v = (const float*)d_in[29];
  const float* wsc  = (const float*)d_in[30];
  const float* bsc  = (const float*)d_in[31];
  const float* bnsg = (const float*)d_in[32];
  const float* bnsb = (const float*)d_in[33];
  const float* bnsm = (const float*)d_in[34];
  const float* bnsv = (const float*)d_in[35];
  const float* wfc  = (const float*)d_in[36];
  const float* bfc  = (const float*)d_in[37];

  float* out_in   = (float*)d_out;                 // x_in  [N,16]
  float* out_out  = (float*)d_out + NN * 16;       // x_out [N,16]
  float* out_self = (float*)d_out + 2 * NN * 16;   // x_self[N,16]

  // workspace carving (~110 MB total)
  char* w = (char*)d_ws;
  size_t off = 0;
  auto take = [&](size_t bytes) { char* pp = w + off; off += (bytes + 255) & ~(size_t)255; return pp; };
  _Float16* xh  = (_Float16*)take((size_t)NN * 128 * 2);   // f16(x)
  _Float16* xh2 = (_Float16*)take((size_t)NN * 64 * 2);    // f16 of layer-1/3 output
  _Float16* Wt  = (_Float16*)take((size_t)2048 * 128 * 2); // transposed weights (max)
  float* h   = (float*)take((size_t)NN * 2048 * 4);        // GEMM output [N, H*C]
  float* es  = (float*)take((size_t)NN * HH * 4);
  float* ed  = (float*)take((size_t)NN * HH * 4);
  float* m   = (float*)take((size_t)NN * HH * 4);
  float* z   = (float*)take((size_t)NN * HH * 4);
  float* p   = (float*)take((size_t)EP * HH * 4);          // per-edge logits/alpha
  float* agg = (float*)take((size_t)NN * 64 * 4);          // aggregation / sc temp
  float* xs  = (float*)take((size_t)NN * 64 * 4);          // x_s / x_t
  float* t1  = (float*)take((size_t)NN * 16 * 4);
  float* t2  = (float*)take((size_t)NN * 16 * 4);

  k_f32_to_f16<<<gdim(NN * 128, 256), 256, 0, stream>>>(xh, x, NN * 128);

  // ---- in-branch: source_to_target
  run_gat_layer(xh, 128, W1, as1, ad1, b1, 64, ei, 0, Wt, h, es, ed, m, z, p, agg, xs, stream);
  k_f32_to_f16<<<gdim(NN * 64, 256), 256, 0, stream>>>(xh2, xs, NN * 64);
  run_gat_layer(xh2, 64, W2, as2, ad2, b2, 16, ei, 0, Wt, h, es, ed, m, z, p, agg, out_in, stream);

  // ---- out-branch: target_to_source
  run_gat_layer(xh, 128, W3, as3, ad3, b3, 64, ei, 1, Wt, h, es, ed, m, z, p, agg, xs, stream);
  k_f32_to_f16<<<gdim(NN * 64, 256), 256, 0, stream>>>(xh2, xs, NN * 64);
  run_gat_layer(xh2, 64, W4, as4, ad4, b4, 16, ei, 1, Wt, h, es, ed, m, z, p, agg, out_out, stream);

  // ---- residual network (x_self)
  const int w16 = NN / 16;  // 625 waves for Nc=16 GEMMs (NT=1)
  k_transpose_to_f16<<<gdim(128 * 16, 256), 256, 0, stream>>>(Wt, rwa, 128, 16);
  k_gemm_wmma<1><<<gdim(w16, 4), 128, 0, stream>>>(xh, Wt, t1, NN, 128, 16);
  k_bias_bn<<<gdim(NN * 16, 256), 256, 0, stream>>>(t1, rba, bn1g, bn1b, bn1m, bn1v, t1, 1, 1);
  k_mm16<<<gdim(NN * 16, 256), 256, 0, stream>>>(t1, rwb, rbb, t2);
  k_bias_bn<<<gdim(NN * 16, 256), 256, 0, stream>>>(t2, rbb, bn2g, bn2b, bn2m, bn2v, t2, 0, 0);
  k_transpose_to_f16<<<gdim(128 * 16, 256), 256, 0, stream>>>(Wt, wsc, 128, 16);
  k_gemm_wmma<1><<<gdim(w16, 4), 128, 0, stream>>>(xh, Wt, agg, NN, 128, 16);
  k_bias_bn<<<gdim(NN * 16, 256), 256, 0, stream>>>(agg, bsc, bnsg, bnsb, bnsm, bnsv, agg, 1, 0);
  k_add_relu<<<gdim(NN * 16, 256), 256, 0, stream>>>(t2, agg, t1, NN * 16);
  k_mm16<<<gdim(NN * 16, 256), 256, 0, stream>>>(t1, wfc, bfc, out_self);
}
